// GCN_32160715112881
// MI455X (gfx1250) — compile-verified, matrix operand order checked by reading, not run on hardware
//
#include <hip/hip_runtime.h>

typedef __bf16 bf16_t;
typedef __attribute__((ext_vector_type(8)))  __bf16 v8bf;
typedef __attribute__((ext_vector_type(16))) __bf16 v16bf;
typedef __attribute__((ext_vector_type(8)))  float  v8f;
typedef __attribute__((ext_vector_type(4)))  float  v4f;

#define NN   12288
#define D    128
#define KNN  20
#define EDG  (NN * (KNN + 1))   // K top-K edges + 1 self-loop edge per row

// ---------------------------------------------------------------------------
// Load a 16-element bf16 vector in CDNA5 WMMA 16-bit operand layout.
// Per ISA 7.12.2 (16-bit A 16x32): lanes 0-15 hold K = {k0..k0+7, k0+16..k0+23},
// lanes 16-31 hold K = {k0+8..k0+15, k0+24..k0+31}.  Two contiguous b128 loads.
// ---------------------------------------------------------------------------
__device__ __forceinline__ v16bf load_tile16(const bf16_t* __restrict__ base,
                                             int row, int k0, int lane) {
  const bf16_t* p = base + (size_t)row * D + k0 + ((lane & 16) ? 8 : 0);
  v8bf lo = *(const v8bf*)(p);
  v8bf hi = *(const v8bf*)(p + 16);
  v16bf r;
#pragma unroll
  for (int i = 0; i < 8; ++i) { r[i] = lo[i]; r[i + 8] = hi[i]; }
  return r;
}

__device__ __forceinline__ v8f wmma_bf16(v16bf a, v16bf b, v8f c) {
  return __builtin_amdgcn_wmma_f32_16x16x32_bf16(false, a, false, b,
                                                 (short)0, c, false, false);
}

// ---------------------------------------------------------------------------
// Row-normalize x; emit bf16 x_norm and bf16 x.
// ---------------------------------------------------------------------------
__global__ __launch_bounds__(128) void prep_x_kernel(const float* __restrict__ x,
                                                     bf16_t* __restrict__ xn,
                                                     bf16_t* __restrict__ xb) {
  __shared__ float red[128];
  int r = blockIdx.x, t = threadIdx.x;
  float v = x[(size_t)r * D + t];
  red[t] = v * v;
  __syncthreads();
#pragma unroll
  for (int s = 64; s > 0; s >>= 1) {
    if (t < s) red[t] += red[t + s];
    __syncthreads();
  }
  float inv = 1.0f / fmaxf(sqrtf(red[0]), 1e-12f);
  xn[(size_t)r * D + t] = (bf16_t)(v * inv);
  xb[(size_t)r * D + t] = (bf16_t)v;
}

__global__ void prep_w_kernel(const float* __restrict__ W1,
                              const float* __restrict__ W2,
                              bf16_t* __restrict__ w1t,
                              bf16_t* __restrict__ w2t) {
  int i = blockIdx.x * blockDim.x + threadIdx.x;
  if (i < 128 * 128) {
    int c = i >> 7, k = i & 127;
    w1t[i] = (bf16_t)W1[k * 128 + c];
  }
  if (i < 64 * 128) {
    int c = i >> 7, k = i & 127;
    w2t[i] = (bf16_t)W2[k * 64 + c];
  }
}

// ---------------------------------------------------------------------------
// Fused Gram + top-K.  One wave owns 32 rows (two A tile-sets, shared B tiles),
// software-pipelined B prefetch; candidate scan reads 4 x ds_load_b128 per tile
// (rotating through each 4-vector) instead of 16 serialized ds_load_b32;
// insertion is branchless under one rare guard, single code copy.
// ---------------------------------------------------------------------------
__global__ __launch_bounds__(32) void sim_topk_kernel(const bf16_t* __restrict__ xn,
                                                      float* __restrict__ vals,
                                                      int*   __restrict__ idxs,
                                                      float* __restrict__ loopw) {
  __shared__ float tile[32][20];       // 20-float row stride: 16B-aligned chunks
  int lane = threadIdx.x;
  int r0 = blockIdx.x * 32;
  int rA = r0 + (lane & 15);
  int rB = r0 + 16 + (lane & 15);

  v16bf a0 = load_tile16(xn, rA, 0, lane);
  v16bf a1 = load_tile16(xn, rA, 32, lane);
  v16bf a2 = load_tile16(xn, rA, 64, lane);
  v16bf a3 = load_tile16(xn, rA, 96, lane);
  v16bf a4 = load_tile16(xn, rB, 0, lane);
  v16bf a5 = load_tile16(xn, rB, 32, lane);
  v16bf a6 = load_tile16(xn, rB, 64, lane);
  v16bf a7 = load_tile16(xn, rB, 96, lane);

  float tv[KNN]; int ti[KNN];
#pragma unroll
  for (int j = 0; j < KNN; ++j) { tv[j] = -2.0f; ti[j] = 0; }

  // prefetch tile 0's B operands
  int cl = lane & 15;
  v16bf b0 = load_tile16(xn, cl, 0, lane);
  v16bf b1 = load_tile16(xn, cl, 32, lane);
  v16bf b2 = load_tile16(xn, cl, 64, lane);
  v16bf b3 = load_tile16(xn, cl, 96, lane);

  for (int ct = 0; ct < NN / 16; ++ct) {
    int c0 = ct * 16;
    // prefetch next tile's B while this tile's WMMAs run
    int ctn = (ct + 1 < NN / 16) ? ct + 1 : 0;
    int cn = ctn * 16 + cl;
    v16bf n0 = load_tile16(xn, cn, 0, lane);
    v16bf n1 = load_tile16(xn, cn, 32, lane);
    v16bf n2 = load_tile16(xn, cn, 64, lane);
    v16bf n3 = load_tile16(xn, cn, 96, lane);

    v8f accA = {};
    v8f accB = {};
    accA = wmma_bf16(a0, b0, accA);
    accB = wmma_bf16(a4, b0, accB);
    accA = wmma_bf16(a1, b1, accA);
    accB = wmma_bf16(a5, b1, accB);
    accA = wmma_bf16(a2, b2, accA);
    accB = wmma_bf16(a6, b2, accB);
    accA = wmma_bf16(a3, b3, accA);
    accB = wmma_bf16(a7, b3, accB);
    b0 = n0; b1 = n1; b2 = n2; b3 = n3;

    int mb = (lane >> 4) * 8, nc = lane & 15;
#pragma unroll
    for (int i = 0; i < 8; ++i) tile[mb + i][nc] = accA[i];
#pragma unroll
    for (int i = 0; i < 8; ++i) tile[16 + mb + i][nc] = accB[i];
    __syncthreads();

    // every lane streams its own row's 16 candidates, 4 at a time (b128)
#pragma unroll 1
    for (int g = 0; g < 4; ++g) {
      v4f cand = *(const v4f*)&tile[lane][g * 4];
#pragma unroll 1
      for (int q = 0; q < 4; ++q) {
        float v = cand[0];
        cand[0] = cand[1]; cand[1] = cand[2]; cand[2] = cand[3];
        if (v > tv[KNN - 1]) {         // rare guard; body is branchless
          float cv = v; int ci = c0 + g * 4 + q;
#pragma unroll
          for (int j = 0; j < KNN; ++j) {
            bool p = cv > tv[j];
            float tf = tv[j]; int si = ti[j];
            tv[j] = p ? cv : tf;
            ti[j] = p ? ci : si;
            cv = p ? tf : cv;
            ci = p ? si : ci;
          }
        }
      }
    }
    __syncthreads();
  }

  int row = r0 + lane;
  bool hs = false;
#pragma unroll
  for (int j = 0; j < KNN; ++j) hs = hs || (ti[j] == row);
  loopw[row] = hs ? 0.0f : 1.0f;       // add_remaining_self_loops(fill=1)
#pragma unroll
  for (int j = 0; j < KNN; ++j) {
    vals[row * KNN + j] = tv[j];
    idxs[row * KNN + j] = ti[j];
  }
}

// ---------------------------------------------------------------------------
// CSC (transpose adjacency) build: count -> scan -> fill.
// ---------------------------------------------------------------------------
__global__ void edge_count_kernel(const int* __restrict__ idxs, int* __restrict__ cnt) {
  int r = blockIdx.x * blockDim.x + threadIdx.x;
  if (r < NN) {
#pragma unroll
    for (int e = 0; e < KNN; ++e) atomicAdd(&cnt[idxs[r * KNN + e]], 1);
    atomicAdd(&cnt[r], 1);             // self-loop edge (weight may be 0)
  }
}

__global__ __launch_bounds__(1024) void scan_kernel(const int* __restrict__ cnt,
                                                    int* __restrict__ off,
                                                    int* __restrict__ cursor) {
  __shared__ int part[1024];
  int t = threadIdx.x;
  int base = t * (NN / 1024);
  int loc[NN / 1024];
  int sum = 0;
#pragma unroll
  for (int i = 0; i < NN / 1024; ++i) { loc[i] = sum; sum += cnt[base + i]; }
  part[t] = sum;
  __syncthreads();
  for (int st = 1; st < 1024; st <<= 1) {
    int v = (t >= st) ? part[t - st] : 0;
    __syncthreads();
    part[t] += v;
    __syncthreads();
  }
  int tb = (t == 0) ? 0 : part[t - 1];
#pragma unroll
  for (int i = 0; i < NN / 1024; ++i) {
    int o = tb + loc[i];
    off[base + i] = o;
    cursor[base + i] = o;
  }
  if (t == 1023) off[NN] = part[1023];
}

__global__ void edge_fill_kernel(const int* __restrict__ idxs,
                                 const float* __restrict__ vals,
                                 const float* __restrict__ lpw,
                                 int* __restrict__ cursor,
                                 int* __restrict__ srcs,
                                 float* __restrict__ wv) {
  int r = blockIdx.x * blockDim.x + threadIdx.x;
  if (r < NN) {
#pragma unroll
    for (int e = 0; e < KNN; ++e) {
      int c = idxs[r * KNN + e];
      int p = atomicAdd(&cursor[c], 1);
      srcs[p] = r;
      wv[p] = vals[r * KNN + e];
    }
    int p = atomicAdd(&cursor[r], 1);
    srcs[p] = r;
    wv[p] = lpw[r];
  }
}

// deg[c] = sum of in-edge weights; dis = rsqrt(deg) (deterministic gather)
__global__ void deg_dis_kernel(const int* __restrict__ off,
                               const float* __restrict__ wv,
                               float* __restrict__ disv) {
  int c = blockIdx.x * blockDim.x + threadIdx.x;
  if (c < NN) {
    float d = 0.0f;
    int s1 = off[c + 1];
    for (int j = off[c]; j < s1; ++j) d += wv[j];
    disv[c] = (d > 0.0f) ? rsqrtf(fmaxf(d, 1e-30f)) : 0.0f;
  }
}

// ---------------------------------------------------------------------------
// Dense transform: 16x16 tile, K=128, bf16 WMMA with fp32 accumulate.
// ---------------------------------------------------------------------------
__global__ __launch_bounds__(32) void gemm16_kernel(const bf16_t* __restrict__ A,
                                                    const bf16_t* __restrict__ BT,
                                                    float* __restrict__ C, int ldc) {
  int lane = threadIdx.x;
  int tm = blockIdx.x * 16, tn = blockIdx.y * 16;
  int ra = tm + (lane & 15), rb = tn + (lane & 15);
  v8f acc = {};
#pragma unroll
  for (int s = 0; s < 4; ++s)
    acc = wmma_bf16(load_tile16(A, ra, 32 * s, lane),
                    load_tile16(BT, rb, 32 * s, lane), acc);
  int m0 = tm + ((lane >> 4) * 8), n0 = tn + (lane & 15);
#pragma unroll
  for (int i = 0; i < 8; ++i) C[(size_t)(m0 + i) * ldc + n0] = acc[i];
}

// Gather aggregation: out[c] = b + sum_in-edges dis[c]*w*dis[r] * hW[r]  (no atomics)
__global__ void gather_agg_kernel(const float* __restrict__ hW,
                                  const int* __restrict__ off,
                                  const int* __restrict__ srcs,
                                  const float* __restrict__ wv,
                                  const float* __restrict__ disv,
                                  const float* __restrict__ bias,
                                  float* __restrict__ out, int F) {
  int c = blockIdx.x, t = threadIdx.x;
  int s1 = off[c + 1];
  float dc = disv[c];
  float acc = bias[t];
  for (int j = off[c]; j < s1; ++j) {
    int r = srcs[j];
    acc += dc * wv[j] * disv[r] * hW[(size_t)r * F + t];
  }
  out[(size_t)c * F + t] = acc;
}

__global__ void relu_cvt_kernel(const float* __restrict__ hacc,
                                bf16_t* __restrict__ hbf) {
  int i = blockIdx.x * blockDim.x + threadIdx.x;
  if (i < NN * 128) hbf[i] = (bf16_t)fmaxf(hacc[i], 0.0f);
}

// ---------------------------------------------------------------------------
extern "C" void kernel_launch(void* const* d_in, const int* in_sizes, int n_in,
                              void* d_out, int out_size, void* d_ws, size_t ws_size,
                              hipStream_t stream) {
  (void)in_sizes; (void)n_in; (void)out_size; (void)ws_size;
  const float* x  = (const float*)d_in[0];
  const float* W1 = (const float*)d_in[1];
  const float* b1 = (const float*)d_in[2];
  const float* W2 = (const float*)d_in[3];
  const float* b2 = (const float*)d_in[4];
  float* out = (float*)d_out;

  char* p = (char*)d_ws;
  auto alloc = [&](size_t bytes) -> char* {
    char* q = p; p += (bytes + 255) & ~(size_t)255; return q;
  };
  bf16_t* xn   = (bf16_t*)alloc((size_t)NN * D * 2);
  bf16_t* xb   = (bf16_t*)alloc((size_t)NN * D * 2);
  bf16_t* w1t  = (bf16_t*)alloc((size_t)128 * 128 * 2);
  bf16_t* w2t  = (bf16_t*)alloc((size_t)64 * 128 * 2);
  float*  vals = (float*)alloc((size_t)NN * KNN * 4);
  int*    idxs = (int*)  alloc((size_t)NN * KNN * 4);
  float*  lpw  = (float*)alloc((size_t)NN * 4);
  int*    cnt  = (int*)  alloc((size_t)NN * 4);
  int*    off  = (int*)  alloc((size_t)(NN + 1) * 4);
  int*    cur  = (int*)  alloc((size_t)NN * 4);
  int*    srcs = (int*)  alloc((size_t)EDG * 4);
  float*  wv   = (float*)alloc((size_t)EDG * 4);
  float*  disv = (float*)alloc((size_t)NN * 4);
  float*  hW   = (float*)alloc((size_t)NN * 128 * 4);
  float*  hacc = (float*)alloc((size_t)NN * 128 * 4);
  bf16_t* hbf  = (bf16_t*)alloc((size_t)NN * 128 * 2);
  float*  hW2  = (float*)alloc((size_t)NN * 64 * 4);

  // Phase 0: precision prep
  prep_x_kernel<<<NN, 128, 0, stream>>>(x, xn, xb);
  prep_w_kernel<<<(128 * 128 + 255) / 256, 256, 0, stream>>>(W1, W2, w1t, w2t);

  // Phase 1: fused Gram + top-20 (WMMA, never materializes adj)
  sim_topk_kernel<<<NN / 32, 32, 0, stream>>>(xn, vals, idxs, lpw);

  // Phase 2: CSC transpose build + gcn_norm
  hipMemsetAsync(cnt, 0, (size_t)NN * 4, stream);
  edge_count_kernel<<<(NN + 255) / 256, 256, 0, stream>>>(idxs, cnt);
  scan_kernel<<<1, 1024, 0, stream>>>(cnt, off, cur);
  edge_fill_kernel<<<(NN + 255) / 256, 256, 0, stream>>>(idxs, vals, lpw, cur, srcs, wv);
  deg_dis_kernel<<<(NN + 255) / 256, 256, 0, stream>>>(off, wv, disv);

  // Phase 3: conv1 = relu(agg(x @ W1) + b1)
  gemm16_kernel<<<dim3(NN / 16, 128 / 16), 32, 0, stream>>>(xb, w1t, hW, 128);
  gather_agg_kernel<<<NN, 128, 0, stream>>>(hW, off, srcs, wv, disv, b1, hacc, 128);
  relu_cvt_kernel<<<(NN * 128 + 255) / 256, 256, 0, stream>>>(hacc, hbf);

  // Phase 4: conv2 = agg(h @ W2) + b2
  gemm16_kernel<<<dim3(NN / 16, 64 / 16), 32, 0, stream>>>(hbf, w2t, hW2, 64);
  gather_agg_kernel<<<NN, 64, 0, stream>>>(hW2, off, srcs, wv, disv, b2, out, 64);
}
